// LSTM_89764816486841
// MI455X (gfx1250) — matrix operand order repaired and checked by reading, not compiled, not run on hardware
//
#include <hip/hip_runtime.h>
#include <hip/hip_bf16.h>
#include <math.h>

// ---------------------------------------------------------------------------
// LSTM: T=512, N=64 batch, F=256 in-features, S=1024 hidden.
// Per step: z[4][64][1024] = [x_t | h] (64x1280) @ Wcomb (1280x4096) + b
//   * one-time: pack Wx/Wh (f32) -> f16 in exact WMMA B-fragment layout
//   * one-time: convert all x -> f16
//   * 512 step kernels: 64 blocks (one per 16-wide hidden tile) x 16 waves
//     = 4 gates x 4-way K-split.  Each wave: 4 M-tiles x 10 k-chunks of
//     v_wmma_f32_16x16x32_f16 (short chains + 4 waves/SIMD to hide L2
//     latency), partials combined via ds_add_f32 into 16 KB LDS, then
//     fused sigmoid/tanh gate math.
//   * H double-buffered across steps (cross-block RAW), C updated in place.
// ---------------------------------------------------------------------------

typedef __attribute__((ext_vector_type(8)))  _Float16 v8h;
typedef __attribute__((ext_vector_type(16))) _Float16 v16h;
typedef __attribute__((ext_vector_type(8)))  float    v8f;

#define T_STEPS 512
#define NB      64
#define FIN     256
#define SH      1024
#define KC_N    40              // 1280 / 32
#define KC_PER  10              // k-chunks per k-split wave
#define JT_N    256             // 4096 / 16

// ws layout (bytes):
#define WPK_BYTES  (JT_N * KC_N * 32 * 16 * 2)          // 10,485,760
#define AX_BYTES   (T_STEPS * NB * FIN * 2)             // 16,777,216
#define H_BYTES    (NB * SH * 2)                        // 131,072

// ---------------------------------------------------------------------------
// Pack Wx[4][256][1024], Wh[4][1024][1024] (f32) into f16 B-fragments.
// B-fragment layout used here (A loads use the same k order): lane holds column n = lane%16,
// K = (lane/16)*16 + e for e in 0..15 (16 contiguous K per lane).
// Packed index: ((jt*KC_N + kc)*32 + lane)*16 + e, jt indexes gate-major
// columns j = jt*16 + n, j in [0,4096), gate = j>>10, s = j&1023.
// ---------------------------------------------------------------------------
__global__ __launch_bounds__(256) void lstm_pack_w(const float* __restrict__ Wx,
                                                   const float* __restrict__ Wh,
                                                   _Float16* __restrict__ Wpk) {
  const int i    = blockIdx.x * 256 + threadIdx.x;   // < 5,242,880
  const int e    = i & 15;
  const int lane = (i >> 4) & 31;
  const int kc   = (i >> 9) % KC_N;
  const int jt   = i / (KC_N * 512);
  const int half = lane >> 4;
  const int n    = lane & 15;
  const int j = jt * 16 + n;
  const int g = j >> 10;
  const int s = j & 1023;
  const int k = kc * 32 + half * 16 + e;
  float w = (k < FIN) ? Wx[((size_t)g * FIN + k) * SH + s]
                      : Wh[((size_t)g * SH + (k - FIN)) * SH + s];
  Wpk[i] = (_Float16)w;
}

__global__ __launch_bounds__(256) void lstm_cvt_x(const float* __restrict__ x,
                                                  _Float16* __restrict__ Ax) {
  const size_t i = (size_t)blockIdx.x * 256 + threadIdx.x;  // < 8,388,608
  Ax[i] = (_Float16)x[i];
}

__global__ __launch_bounds__(256) void lstm_init_state(_Float16* __restrict__ H0,
                                                       float* __restrict__ C) {
  const int i = blockIdx.x * 256 + threadIdx.x;             // < 65,536
  H0[i] = (_Float16)0.0f;
  C[i]  = 0.0f;
}

// Combine two v8h loads into one v16h WMMA operand.
static __device__ __forceinline__ v16h cat16(v8h lo, v8h hi) {
  return __builtin_shufflevector(lo, hi, 0, 1, 2, 3, 4, 5, 6, 7,
                                 8, 9, 10, 11, 12, 13, 14, 15);
}

// ---------------------------------------------------------------------------
// One LSTM time step.  grid = 64 blocks (st = hidden tile), block = 512
// threads = 16 waves.  Wave w: gate g = w&3, k-split ks = w>>2 handles
// k-chunks [ks*10, ks*10+10) for hidden columns [st*16, st*16+16) over all
// 64 batch rows (4 M-tiles).
// ---------------------------------------------------------------------------
__global__ __launch_bounds__(512) void lstm_step(
    const _Float16* __restrict__ Ax,     // [NB][FIN] slice for this t
    const _Float16* __restrict__ Hin,    // [NB][SH]  (f16, previous h)
    const _Float16* __restrict__ Wpk,    // packed weights
    const float*    __restrict__ bias4,  // [4][SH]
    float*          __restrict__ C,      // [NB][SH]  in/out cell state
    _Float16*       __restrict__ Hout,   // [NB][SH]  next h (f16)
    float*          __restrict__ out_t)  // [NB][SH]  hs[t]
{
  const int st   = blockIdx.x;             // 0..63
  // Wave-uniform scalars: force into SGPRs so control flow stays uniform
  // (no EXEC masking around WMMA, scalar loop counters).
  const int w    = __builtin_amdgcn_readfirstlane((int)(threadIdx.x >> 5));
  const int g    = w & 3;                  // gate
  const int ks   = w >> 2;                 // k-split 0..3
  const int lane = threadIdx.x & 31;
  const int half = lane >> 4;
  const int n    = lane & 15;
  const int tid  = threadIdx.x;

  __shared__ float zsh[4][NB][16];         // 16 KB: z for all gates of tile
  float* zflat = &zsh[0][0][0];

  // Zero the z accumulation buffer (4096 floats, 512 threads x 8).
#pragma unroll
  for (int j = 0; j < 8; ++j) zflat[j * 512 + tid] = 0.0f;
  __syncthreads();

  v8f acc[4];
#pragma unroll
  for (int mt = 0; mt < 4; ++mt)
    acc[mt] = (v8f){0.f, 0.f, 0.f, 0.f, 0.f, 0.f, 0.f, 0.f};

  const int jt  = g * 64 + st;
  const int kc0 = ks * KC_PER;             // uniform
  const int kc1 = kc0 + KC_PER;            // uniform
  const int xEnd   = (kc1 < 8) ? kc1 : 8;  // x-part upper bound
  const int hStart = (kc0 > 8) ? kc0 : 8;  // h-part lower bound

  const _Float16* wp = Wpk + ((size_t)jt * KC_N + kc0) * 512 + lane * 16;

  // ---- x-part: kc in [kc0, xEnd), A stride = FIN (constant) ----
  {
    const _Float16* ap0 = Ax + (size_t)n * FIN + half * 16 + kc0 * 32;
    for (int kc = kc0; kc < xEnd; ++kc) {
      v16h bf = cat16(*(const v8h*)(wp), *(const v8h*)(wp + 8));
      wp += 512;
      __builtin_prefetch((const void*)wp, 0, 3);
#pragma unroll
      for (int mt = 0; mt < 4; ++mt) {
        const _Float16* ap = ap0 + (size_t)mt * 16 * FIN;
        v16h af = cat16(*(const v8h*)(ap), *(const v8h*)(ap + 8));
        acc[mt] = __builtin_amdgcn_wmma_f32_16x16x32_f16(
            false, af, false, bf, (short)0, acc[mt], false, false);
      }
      ap0 += 32;
    }
  }

  // ---- h-part: kc in [hStart, kc1), A stride = SH (constant) ----
  {
    const _Float16* ap0 = Hin + (size_t)n * SH + half * 16 + (hStart - 8) * 32;
    for (int kc = hStart; kc < kc1; ++kc) {
      v16h bf = cat16(*(const v8h*)(wp), *(const v8h*)(wp + 8));
      wp += 512;
      __builtin_prefetch((const void*)wp, 0, 3);
#pragma unroll
      for (int mt = 0; mt < 4; ++mt) {
        const _Float16* ap = ap0 + (size_t)mt * 16 * SH;
        v16h af = cat16(*(const v8h*)(ap), *(const v8h*)(ap + 8));
        acc[mt] = __builtin_amdgcn_wmma_f32_16x16x32_f16(
            false, af, false, bf, (short)0, acc[mt], false, false);
      }
      ap0 += 32;
    }
  }

  // Combine k-split partials: C/D layout puts VGPR r of lane at row
  // M = r + 8*half, col = lane%16.  ds_add_f32 accumulation into LDS.
#pragma unroll
  for (int mt = 0; mt < 4; ++mt)
#pragma unroll
    for (int r = 0; r < 8; ++r)
      (void)atomicAdd(&zsh[g][mt * 16 + half * 8 + r][n], acc[mt][r]);

  __syncthreads();

  // Gate fusion: 1024 elements (64 batch x 16 cols), 2 per thread, coalesced.
#pragma unroll
  for (int e = 0; e < 2; ++e) {
    const int idx = e * 512 + tid;      // 0..1023
    const int nb  = idx >> 4;
    const int sl  = idx & 15;
    const int s   = st * 16 + sl;
    const float zi = zsh[0][nb][sl] + bias4[0 * SH + s];
    const float zf = zsh[1][nb][sl] + bias4[1 * SH + s];
    const float zg = zsh[2][nb][sl] + bias4[2 * SH + s];
    const float zo = zsh[3][nb][sl] + bias4[3 * SH + s];
    const float ig = 1.0f / (1.0f + __expf(-zi));
    const float fg = 1.0f / (1.0f + __expf(-zf));
    const float gg = tanhf(zg);
    const float og = 1.0f / (1.0f + __expf(-zo));
    const size_t gi = (size_t)nb * SH + s;
    const float c = fg * C[gi] + ig * gg;
    C[gi] = c;
    const float h = og * tanhf(c);
    out_t[gi] = h;
    Hout[gi]  = (_Float16)h;
  }
}

// ---------------------------------------------------------------------------
extern "C" void kernel_launch(void* const* d_in, const int* in_sizes, int n_in,
                              void* d_out, int out_size, void* d_ws, size_t ws_size,
                              hipStream_t stream) {
  (void)in_sizes; (void)n_in; (void)out_size; (void)ws_size;
  const float* x  = (const float*)d_in[0];   // [512][64][256]
  const float* Wx = (const float*)d_in[1];   // [4][256][1024]
  const float* Wh = (const float*)d_in[2];   // [4][1024][1024]
  const float* b  = (const float*)d_in[3];   // [4][1024]
  float* out = (float*)d_out;                // [512][64][1024]

  char* ws = (char*)d_ws;
  _Float16* Wpk = (_Float16*)(ws);
  _Float16* Ax  = (_Float16*)(ws + WPK_BYTES);
  _Float16* H0  = (_Float16*)(ws + WPK_BYTES + AX_BYTES);
  _Float16* H1  = (_Float16*)(ws + WPK_BYTES + AX_BYTES + H_BYTES);
  float*    C   = (float*)   (ws + WPK_BYTES + AX_BYTES + 2 * H_BYTES);

  lstm_pack_w<<<(JT_N * KC_N * 512) / 256, 256, 0, stream>>>(Wx, Wh, Wpk);
  lstm_cvt_x<<<(T_STEPS * NB * FIN) / 256, 256, 0, stream>>>(x, Ax);
  lstm_init_state<<<(NB * SH) / 256, 256, 0, stream>>>(H0, C);

  for (int t = 0; t < T_STEPS; ++t) {
    const _Float16* Hin  = (t & 1) ? H1 : H0;
    _Float16*       Hout = (t & 1) ? H0 : H1;
    lstm_step<<<64, 512, 0, stream>>>(Ax + (size_t)t * NB * FIN, Hin, Wpk, b,
                                      C, Hout, out + (size_t)t * NB * SH);
  }
}
